// s4MambaModule_24154896073579
// MI455X (gfx1250) — matrix operand order repaired and checked by reading, not compiled, not run on hardware
//
#include <hip/hip_runtime.h>
#include <hip/hip_bf16.h>

typedef __bf16        v16bf __attribute__((ext_vector_type(16)));
typedef float         v8f   __attribute__((ext_vector_type(8)));
typedef unsigned int  v4u   __attribute__((ext_vector_type(4)));
typedef unsigned int  u32x4 __attribute__((ext_vector_type(4)));
typedef int           i32x4 __attribute__((ext_vector_type(4)));
typedef int           i32x8 __attribute__((ext_vector_type(8)));

static constexpr int Bn  = 4;
static constexpr int Ln  = 4096;
static constexpr int DM  = 1024;
static constexpr int Hn  = 2048;
static constexpr int NST = 8;
static constexpr int LDSS = 48;   // LDS row stride (halves): 32 data + 16 pad -> 96B rows, 16B aligned

#if defined(__HIP_DEVICE_COMPILE__) && defined(__gfx1250__) && \
    __has_builtin(__builtin_amdgcn_tensor_load_to_lds) &&      \
    __has_builtin(__builtin_amdgcn_s_wait_tensorcnt)
#define HAVE_TDM 1
#else
#define HAVE_TDM 0
#endif

// ---------------------------------------------------------------- conversions
__global__ void f32_to_bf16_k(const float* __restrict__ in, __bf16* __restrict__ out, int n) {
  int i = blockIdx.x * blockDim.x + threadIdx.x;
  int stride = gridDim.x * blockDim.x;
  for (; i < n; i += stride) out[i] = (__bf16)in[i];
}

#if HAVE_TDM
// Flat pointer to shared memory: addr[31:0] is the LDS byte offset (aperture layout).
__device__ inline unsigned lds_off(const void* p) {
  return (unsigned)(unsigned long long)p;
}

// Issue one TDM load of a (tile_dim1=128 rows) x (tile_dim0=32 elems) bf16 tile into LDS,
// padding 8 DWORDs after every 16 DWORDs so LDS rows land at LDSS=48-half stride.
__device__ inline void tdm_load_tile(unsigned lds_byte, const __bf16* gtile,
                                     unsigned strideElems, unsigned dim0rem, unsigned dim1rem) {
  unsigned long long ga = (unsigned long long)(const void*)gtile;
  u32x4 g0;
  g0[0] = 1u;                                              // count=1, user mode, no gather
  g0[1] = lds_byte;                                        // lds_addr
  g0[2] = (unsigned)(ga & 0xFFFFFFFFu);                    // global_addr[31:0]  (bits 95:64)
  g0[3] = (unsigned)((ga >> 32) & 0x01FFFFFFu) | (2u << 30); // addr[56:32] | type=2
  i32x8 g1;
  g1[0] = (1 << 16)                                        // data_size = 2 bytes
        | (1 << 20)                                        // pad_enable
        | (3 << 22)                                        // pad_interval: 16 DWORDs (one 64B row)
        | (7 << 25);                                       // pad_amount : 8 DWORDs (32B)
  g1[1] = (int)((dim0rem & 0xFFFFu) << 16);                // atomic_barrier=0 | tensor_dim0[15:0]
  g1[2] = (int)((dim0rem >> 16) & 0xFFFFu)                 // tensor_dim0[31:16]
        | (int)((dim1rem & 0xFFFFu) << 16);                // tensor_dim1[15:0]
  g1[3] = (int)((dim1rem >> 16) & 0xFFFFu) | (32 << 16);   // tensor_dim1[31:16] | tile_dim0=32
  g1[4] = 128;                                             // tile_dim1=128, tile_dim2=0
  g1[5] = (int)strideElems;                                // tensor_dim0_stride[31:0]
  g1[6] = 0;                                               // stride[47:32]=0, dim1_stride lo=0
  g1[7] = 0;
  const i32x4 gz = {0, 0, 0, 0};
#if __clang_major__ >= 23
  const i32x8 gz8 = {0, 0, 0, 0, 0, 0, 0, 0};
  __builtin_amdgcn_tensor_load_to_lds(g0, g1, gz, gz, gz8, 0);
#else
  __builtin_amdgcn_tensor_load_to_lds(g0, g1, gz, gz, 0);
#endif
}
#endif  // HAVE_TDM

// ---------------------------------------------------------------- WMMA GEMM
// C[M,N] = A[M,K] * Bt[N,K]^T   (A, Bt row-major bf16; f32 accumulate)
// MODE 0: columns n <  Hn -> x transposed to (B,Hn,Ln) f32
//         columns n >= Hn -> silu -> zs transposed to (B,Hn,Ln) bf16   (block-uniform split)
// MODE 1: plain f32 store, row-major M x N
template <int MODE>
__global__ __launch_bounds__(256) void gemm_wmma_bf16(
    const __bf16* __restrict__ A, const __bf16* __restrict__ Bt,
    float* __restrict__ outF, __bf16* __restrict__ outZ,
    int M, int N, int K) {
  __shared__ __bf16 lA[2][128 * LDSS];
  __shared__ __bf16 lB[2][128 * LDSS];

  const int tid    = threadIdx.x;
  const int blockM = blockIdx.y * 128;
  const int blockN = blockIdx.x * 128;
  const int lane   = tid & 31;
  const int wv     = tid >> 5;     // 8 waves
  const int waveM  = wv >> 2;      // 0..1  -> 64 rows each
  const int waveN  = wv & 3;       // 0..3  -> 32 cols each
  const int lm     = lane & 15;
  const int lhi    = lane >> 4;

  const v8f vzero = {0.f, 0.f, 0.f, 0.f, 0.f, 0.f, 0.f, 0.f};
  v8f acc[4][2];
#pragma unroll
  for (int i = 0; i < 4; ++i)
#pragma unroll
    for (int j = 0; j < 2; ++j) acc[i][j] = vzero;

#if HAVE_TDM
  // ---- TDM double-buffered pipeline: wave 0 drives the Tensor Data Mover
  const int nk = K / 32;
  if (wv == 0) {
    tdm_load_tile(lds_off(&lA[0][0]), A  + (size_t)blockM * K, K, K, M - blockM);
    tdm_load_tile(lds_off(&lB[0][0]), Bt + (size_t)blockN * K, K, K, N - blockN);
  }
  for (int k = 0; k < nk; ++k) {
    const int cur = k & 1;
    if (wv == 0) {
      if (k + 1 < nk) {
        const int k0 = (k + 1) * 32;
        tdm_load_tile(lds_off(&lA[cur ^ 1][0]), A  + (size_t)blockM * K + k0, K, K - k0, M - blockM);
        tdm_load_tile(lds_off(&lB[cur ^ 1][0]), Bt + (size_t)blockN * K + k0, K, K - k0, N - blockN);
        __builtin_amdgcn_s_wait_tensorcnt(2);  // tile k done; only tile k+1 outstanding
      } else {
        __builtin_amdgcn_s_wait_tensorcnt(0);
      }
    }
    __syncthreads();   // tile k visible to all waves
    const __bf16* tA = &lA[cur][0];
    const __bf16* tB = &lB[cur][0];

    v16bf af[4];
#pragma unroll
    for (int fm = 0; fm < 4; ++fm) {
      int m = waveM * 64 + fm * 16 + lm, kb = lhi * 16;
#pragma unroll
      for (int j = 0; j < 8; ++j) {
        af[fm][j]     = tA[m * LDSS + kb + j];
        af[fm][j + 8] = tA[m * LDSS + kb + 8 + j];
      }
    }
    v16bf bfv[2];
#pragma unroll
    for (int fn = 0; fn < 2; ++fn) {
      int n = waveN * 32 + fn * 16 + lm, kq = lhi * 16;
#pragma unroll
      for (int j = 0; j < 16; ++j) bfv[fn][j] = tB[n * LDSS + kq + j];
    }
#pragma unroll
    for (int fm = 0; fm < 4; ++fm)
#pragma unroll
      for (int fn = 0; fn < 2; ++fn)
        acc[fm][fn] = __builtin_amdgcn_wmma_f32_16x16x32_bf16(
            false, af[fm], false, bfv[fn], (short)0, acc[fm][fn], false, false);
    __syncthreads();   // all reads of buf[cur] done before it is re-targeted
  }
#else
  // ---- fallback: synchronous per-thread staging
  for (int k0 = 0; k0 < K; k0 += 32) {
#pragma unroll
    for (int i = 0; i < 2; ++i) {
      int c = tid + i * 256, row = c >> 2, ko = (c & 3) << 3;
      *(v4u*)(&lA[0][row * LDSS + ko]) = *(const v4u*)(A  + (size_t)(blockM + row) * K + k0 + ko);
      *(v4u*)(&lB[0][row * LDSS + ko]) = *(const v4u*)(Bt + (size_t)(blockN + row) * K + k0 + ko);
    }
    __syncthreads();
    v16bf af[4];
#pragma unroll
    for (int fm = 0; fm < 4; ++fm) {
      int m = waveM * 64 + fm * 16 + lm, kb = lhi * 16;
#pragma unroll
      for (int j = 0; j < 8; ++j) {
        af[fm][j]     = lA[0][m * LDSS + kb + j];
        af[fm][j + 8] = lA[0][m * LDSS + kb + 8 + j];
      }
    }
    v16bf bfv[2];
#pragma unroll
    for (int fn = 0; fn < 2; ++fn) {
      int n = waveN * 32 + fn * 16 + lm, kq = lhi * 16;
#pragma unroll
      for (int j = 0; j < 16; ++j) bfv[fn][j] = lB[0][n * LDSS + kq + j];
    }
#pragma unroll
    for (int fm = 0; fm < 4; ++fm)
#pragma unroll
      for (int fn = 0; fn < 2; ++fn)
        acc[fm][fn] = __builtin_amdgcn_wmma_f32_16x16x32_bf16(
            false, af[fm], false, bfv[fn], (short)0, acc[fm][fn], false, false);
    __syncthreads();
  }
#endif

  // ---- epilogue: C/D layout — VGPR r: lanes 0-15 M=r, lanes 16-31 M=8+r, N=lane%16
  const bool isX = (MODE == 0) && (blockN < Hn);   // block-uniform x/z split
#pragma unroll
  for (int fm = 0; fm < 4; ++fm) {
#pragma unroll
    for (int fn = 0; fn < 2; ++fn) {
#pragma unroll
      for (int r = 0; r < 8; ++r) {
        int m = blockM + waveM * 64 + fm * 16 + lhi * 8 + r;
        int n = blockN + waveN * 32 + fn * 16 + lm;
        float v = acc[fm][fn][r];
        if (MODE == 1) {
          outF[(size_t)m * N + n] = v;
        } else {
          int b = m >> 12;          // / Ln
          int l = m & (Ln - 1);
          if (isX) {
            outF[((size_t)b * Hn + n) * Ln + l] = v;
          } else {
            float s = v / (1.0f + __expf(-v));   // silu(z)
            outZ[((size_t)b * Hn + (n - Hn)) * Ln + l] = (__bf16)s;
          }
        }
      }
    }
  }
}

// ---------------------------------------------------------------- fused conv + SiLU + S4D chunked scan + gate
// One wave per (b,h); each of 32 lanes owns a 128-step chunk of L.
// Hillis-Steele cross-lane scan with complex geometric weights F = w^128 composes carries.
__global__ __launch_bounds__(256) void s4d_scan_gate(
    const float* __restrict__ x,       // (B,Hn,Ln) raw conv input
    const __bf16* __restrict__ zs,     // (B,Hn,Ln) silu(z)
    const float* __restrict__ conv_w,  // (Hn,4)
    const float* __restrict__ conv_b,  // (Hn)
    const float* __restrict__ log_dt,  // (Hn)
    const float* __restrict__ log_A_real,  // (Hn,NST)
    const float* __restrict__ A_imag,      // (Hn,NST)
    const float* __restrict__ C_re,        // (Hn,NST)
    const float* __restrict__ C_im,        // (Hn,NST)
    const float* __restrict__ Dvec,        // (Hn)
    __bf16* __restrict__ yg)           // (B,Ln,Hn) gated output, GEMM2 A-matrix
{
  const int lane = threadIdx.x & 31;
  const int wv   = threadIdx.x >> 5;
  const int p    = blockIdx.x * 8 + wv;   // flat (b*Hn + h)
  const int b    = p >> 11;               // / Hn
  const int h    = p & (Hn - 1);
  const int CH   = Ln / 32;               // 128

  const float cw0 = conv_w[h * 4 + 0], cw1 = conv_w[h * 4 + 1];
  const float cw2 = conv_w[h * 4 + 2], cw3 = conv_w[h * 4 + 3];
  const float cb  = conv_b[h];
  const float Dh  = Dvec[h];
  const float dt  = __expf(log_dt[h]);

  float wr[NST], wi[NST], c2r[NST], c2i[NST], Fr[NST], Fi[NST];
#pragma unroll
  for (int n = 0; n < NST; ++n) {
    float Ar = -__expf(log_A_real[h * NST + n]);
    float Ai = A_imag[h * NST + n];
    float dr = dt * Ar, di = dt * Ai;
    float er = __expf(dr);
    float wrn = er * __cosf(di), win = er * __sinf(di);   // w = exp(dt*A)
    wr[n] = wrn; wi[n] = win;
    float e1r = wrn - 1.0f, e1i = win;                    // exp(dtA) - 1
    float cr = C_re[h * NST + n], ci = C_im[h * NST + n];
    float nr = cr * e1r - ci * e1i, ni = cr * e1i + ci * e1r;
    float inv = 1.0f / (Ar * Ar + Ai * Ai);
    c2r[n] = 2.0f * (nr * Ar + ni * Ai) * inv;            // 2*Ct = 2*C*(e^dtA-1)/A
    c2i[n] = 2.0f * (ni * Ar - nr * Ai) * inv;
    float epr = __expf((float)CH * dr);                   // F = w^CH
    float pai = (float)CH * di;
    Fr[n] = epr * __cosf(pai);
    Fi[n] = epr * __sinf(pai);
  }

  const size_t rowBase = ((size_t)b * Hn + h) * Ln;
  const float* xp = x + rowBase + lane * CH;

  // ---- phase 1: local chunk scan (states only; conv window seeded from prev chunk)
  float Sr[NST], Si[NST];
#pragma unroll
  for (int n = 0; n < NST; ++n) { Sr[n] = 0.f; Si[n] = 0.f; }
  {
    float xm1 = (lane == 0) ? 0.f : xp[-1];
    float xm2 = (lane == 0) ? 0.f : xp[-2];
    float xm3 = (lane == 0) ? 0.f : xp[-3];
    for (int t = 0; t < CH; ++t) {
      float xc = xp[t];
      float v  = cw0 * xm3 + cw1 * xm2 + cw2 * xm1 + cw3 * xc + cb;
      xm3 = xm2; xm2 = xm1; xm1 = xc;
      float xh = v / (1.0f + __expf(-v));   // silu
#pragma unroll
      for (int n = 0; n < NST; ++n) {
        float tr = wr[n] * Sr[n] - wi[n] * Si[n] + xh;
        float ti = wr[n] * Si[n] + wi[n] * Sr[n];
        Sr[n] = tr; Si[n] = ti;
      }
    }
  }

  // ---- phase 2: inclusive cross-lane scan  S_i = sum_{j<=i} F^(i-j) S_j
#pragma unroll
  for (int s = 1; s < 32; s <<= 1) {
#pragma unroll
    for (int n = 0; n < NST; ++n) {
      float orr = __shfl_up(Sr[n], (unsigned)s, 32);
      float oii = __shfl_up(Si[n], (unsigned)s, 32);
      if (lane >= s) {
        Sr[n] += Fr[n] * orr - Fi[n] * oii;
        Si[n] += Fr[n] * oii + Fi[n] * orr;
      }
    }
#pragma unroll
    for (int n = 0; n < NST; ++n) {   // F <- F*F (complex)
      float fr = Fr[n] * Fr[n] - Fi[n] * Fi[n];
      float fi = 2.0f * Fr[n] * Fi[n];
      Fr[n] = fr; Fi[n] = fi;
    }
  }
  // carry into this chunk = inclusive value of lane-1 (true state at end of prev chunk)
#pragma unroll
  for (int n = 0; n < NST; ++n) {
    float cr_ = __shfl_up(Sr[n], 1u, 32);
    float ci_ = __shfl_up(Si[n], 1u, 32);
    Sr[n] = (lane == 0) ? 0.f : cr_;
    Si[n] = (lane == 0) ? 0.f : ci_;
  }

  // ---- phase 3: rerun with carry, emit gated output in (B,Ln,Hn) layout
  const __bf16* zp = zs + rowBase + lane * CH;
  __bf16* yp = yg + ((size_t)b * Ln + lane * CH) * Hn + h;
  {
    float xm1 = (lane == 0) ? 0.f : xp[-1];
    float xm2 = (lane == 0) ? 0.f : xp[-2];
    float xm3 = (lane == 0) ? 0.f : xp[-3];
    for (int t = 0; t < CH; ++t) {
      float xc = xp[t];
      float v  = cw0 * xm3 + cw1 * xm2 + cw2 * xm1 + cw3 * xc + cb;
      xm3 = xm2; xm2 = xm1; xm1 = xc;
      float xh = v / (1.0f + __expf(-v));
      float acc = Dh * xh;
#pragma unroll
      for (int n = 0; n < NST; ++n) {
        float tr = wr[n] * Sr[n] - wi[n] * Si[n] + xh;
        float ti = wr[n] * Si[n] + wi[n] * Sr[n];
        Sr[n] = tr; Si[n] = ti;
        acc += c2r[n] * tr - c2i[n] * ti;   // 2*Re(Ct * S)
      }
      float zg = (float)zp[t];
      yp[(size_t)t * Hn] = (__bf16)(acc * zg);
    }
  }
}

// ---------------------------------------------------------------- launch
extern "C" void kernel_launch(void* const* d_in, const int* in_sizes, int n_in,
                              void* d_out, int out_size, void* d_ws, size_t ws_size,
                              hipStream_t stream) {
  (void)in_sizes; (void)n_in; (void)out_size; (void)ws_size;
  const float* hs     = (const float*)d_in[0];
  const float* W_in   = (const float*)d_in[1];
  const float* conv_w = (const float*)d_in[2];
  const float* conv_b = (const float*)d_in[3];
  const float* log_dt = (const float*)d_in[4];
  const float* lAreal = (const float*)d_in[5];
  const float* A_imag = (const float*)d_in[6];
  const float* C_re   = (const float*)d_in[7];
  const float* C_im   = (const float*)d_in[8];
  const float* Dv     = (const float*)d_in[9];
  const float* W_out  = (const float*)d_in[10];
  float* out = (float*)d_out;

  char* ws = (char*)d_ws;
  __bf16* hs_bf   = (__bf16*)(ws);                    // 32 MB  (B,L,DM) bf16
  __bf16* win_bf  = (__bf16*)(ws + 33554432ull);      //  8 MB  (2H,DM)  bf16
  __bf16* wout_bf = (__bf16*)(ws + 41943040ull);      //  4 MB  (DM,H)   bf16
  float*  x_t     = (float*) (ws + 46137344ull);      // 128 MB (B,H,L)  f32
  __bf16* zs_t    = (__bf16*)(ws + 180355072ull);     // 64 MB  (B,H,L)  bf16 silu(z)
  __bf16* yg_bf   = (__bf16*)(ws + 213909504ull);     // 64 MB  (B,L,H)  bf16 gated y

  f32_to_bf16_k<<<2048, 256, 0, stream>>>(hs,    hs_bf,   Bn * Ln * DM);
  f32_to_bf16_k<<<1024, 256, 0, stream>>>(W_in,  win_bf,  2 * Hn * DM);
  f32_to_bf16_k<<<512,  256, 0, stream>>>(W_out, wout_bf, DM * Hn);

  // in_proj: (16384 x 1024) x (1024 x 4096); fused split/SiLU/transpose epilogue
  dim3 g1(4096 / 128, 16384 / 128);
  gemm_wmma_bf16<0><<<g1, 256, 0, stream>>>(hs_bf, win_bf, x_t, zs_t,
                                            Bn * Ln, 2 * Hn, DM);

  // conv + SiLU + diagonal-SSM chunked scan + gate  (one wave per (b,h))
  s4d_scan_gate<<<(Bn * Hn) / 8, 256, 0, stream>>>(
      x_t, zs_t, conv_w, conv_b, log_dt, lAreal, A_imag, C_re, C_im, Dv, yg_bf);

  // out_proj: (16384 x 2048) x (2048 x 1024) -> d_out (B,L,DM) f32
  dim3 g2(1024 / 128, 16384 / 128);
  gemm_wmma_bf16<1><<<g2, 256, 0, stream>>>(yg_bf, wout_bf, out, nullptr,
                                            Bn * Ln, DM, Hn);
}
